// ParallelRNNExtractor_45251775430702
// MI455X (gfx1250) — matrix operand order repaired and checked by reading, not compiled, not run on hardware
//
#include <hip/hip_runtime.h>
#include <hip/hip_bf16.h>

// ---------------------------------------------------------------------------
// Fused multi-process GRU scan for MI455X (gfx1250, wave32, WMMA).
//
// One wave32 owns a 16-row batch tile of one process and runs the whole
// T=256 recurrence with h kept in registers (WMMA C/D layout) and W_hh kept
// in registers as f16 WMMA B-layout. Per step: 6x v_wmma_f32_16x16x32_f16
// (full K=H=32 reduction per instruction), gate math in D-layout, and a
// register->LDS->register transpose to rebuild the f16 A operand.
// ---------------------------------------------------------------------------

typedef __attribute__((ext_vector_type(16))) _Float16 v16h;
typedef __attribute__((ext_vector_type(8)))  _Float16 v8h;
typedef __attribute__((ext_vector_type(8)))  float    v8f;

#define B_TOT 8192
#define T_LEN 256
#define P_N   16
#define H_N   32
#define G3    96   // 3*H

__device__ __forceinline__ float fast_sigmoid(float x) {
    // 1 / (1 + 2^(-x*log2e))  -> v_exp_f32 + v_rcp_f32
    float e = __builtin_amdgcn_exp2f(-x * 1.4426950408889634f);
    return __builtin_amdgcn_rcpf(1.0f + e);
}

__device__ __forceinline__ float fast_tanh(float x) {
    // tanh(x) = 2*sigmoid(2x) - 1
    float e = __builtin_amdgcn_exp2f(-x * 2.8853900817779268f);
    return fmaf(2.0f, __builtin_amdgcn_rcpf(1.0f + e), -1.0f);
}

__launch_bounds__(256, 1)
__global__ void gru16_wmma_kernel(const float* __restrict__ x,
                                  const float* __restrict__ W_ih,
                                  const float* __restrict__ W_hh,
                                  const float* __restrict__ b_ih,
                                  const float* __restrict__ b_hh,
                                  float* __restrict__ out)
{
    // per-wave 16x32 f16 transpose scratch (1 KB per wave, 8 waves / block)
    __shared__ __align__(16) _Float16 hbuf[8][16][32];

    const int tid  = threadIdx.x;
    const int wave = tid >> 5;
    const int lane = tid & 31;
    const int n    = lane & 15;   // column within a 16-wide tile / A-row
    const int hi   = lane >> 4;   // lane-half selector

    const int gw = blockIdx.x * 8 + wave;     // global wave id
    const int p  = gw & (P_N - 1);            // process id
    const int b0 = (gw >> 4) * 16;            // batch tile base row

    // ---- preload weights into registers (once) ----------------------------
    // B operand (32x16), contiguous K order as in A: VGPR i holds packed K = 2i,2i+1 (+16 for
    // lanes 16-31); N = lane%16.
    v16h Bm[6];
    float wih[6], bihv[6], bhhv[6];
    #pragma unroll
    for (int t = 0; t < 6; ++t) {
        const int g = 16 * t + n;                         // gate column 0..95
        const float* wrow = W_hh + ((size_t)p * G3 + g) * H_N;
        #pragma unroll
        for (int i = 0; i < 8; ++i) {
            const int k = 2 * i + 16 * hi;
            Bm[t][2 * i]     = (_Float16)wrow[k];
            Bm[t][2 * i + 1] = (_Float16)wrow[k + 1];
        }
        wih[t]  = W_ih[p * G3 + g];
        bihv[t] = b_ih[p * G3 + g];
        bhhv[t] = b_hh[p * G3 + g];
    }

    // h state, f32, WMMA C/D layout: hst[half][r] -> row (r + 8*hi), col 16*half + n
    v8f hst[2];
    #pragma unroll
    for (int i = 0; i < 8; ++i) { hst[0][i] = 0.0f; hst[1][i] = 0.0f; }

    // A operand (f16 A-layout of current h); h0 == 0
    v16h Ah;
    #pragma unroll
    for (int i = 0; i < 16; ++i) Ah[i] = (_Float16)0.0f;

    _Float16* myh = &hbuf[wave][0][0];
    // ds_bpermute byte-index base: +32B for lanes 16-31 (rows 8..15)
    const int idx_base = (lane & 16) << 1;

    // x[b, t, p]: lane streams row b0+n (lanes 16-31 duplicate rows 0..15)
    const float* xptr = x + (size_t)(b0 + n) * T_LEN * P_N + p;

    for (int t = 0; t < T_LEN; ++t) {
        // -------- stream x_t (read-once: non-temporal), prefetch ahead ----
        float xv = __builtin_nontemporal_load(xptr + (size_t)t * P_N);
        if (t + 16 < T_LEN)
            __builtin_prefetch(xptr + (size_t)(t + 16) * P_N, 0, 0);

        // per-row broadcast: xb[r] = x for batch row (r + 8*hi)
        float xb[8];
        #pragma unroll
        for (int r = 0; r < 8; ++r) {
            int v = __builtin_amdgcn_ds_bpermute(idx_base + 4 * r,
                                                 __float_as_int(xv));
            xb[r] = __int_as_float(v);
        }

        // -------- gh = h @ W_hh^T + b_hh : 6x WMMA f32_16x16x32_f16 -------
        v8f d[6];
        #pragma unroll
        for (int tt = 0; tt < 6; ++tt) {
            v8f c;
            #pragma unroll
            for (int i = 0; i < 8; ++i) c[i] = bhhv[tt];   // column bias
            d[tt] = __builtin_amdgcn_wmma_f32_16x16x32_f16(
                        false, Ah, false, Bm[tt], (short)0, c, false, false);
        }

        // -------- gates + state update (D-layout, pure VALU) --------------
        #pragma unroll
        for (int half = 0; half < 2; ++half) {
            #pragma unroll
            for (int r = 0; r < 8; ++r) {
                float gi_r = fmaf(xb[r], wih[half],     bihv[half]);
                float gi_z = fmaf(xb[r], wih[2 + half], bihv[2 + half]);
                float gi_n = fmaf(xb[r], wih[4 + half], bihv[4 + half]);
                float rg = fast_sigmoid(gi_r + d[half][r]);
                float zg = fast_sigmoid(gi_z + d[2 + half][r]);
                float ng = fast_tanh(fmaf(rg, d[4 + half][r], gi_n));
                float hn = fmaf(zg, hst[half][r] - ng, ng); // (1-z)*n + z*h
                hst[half][r] = hn;
                // transpose staging: row m = r + 8*hi, col j = 16*half + n
                myh[(r + 8 * hi) * 32 + 16 * half + n] = (_Float16)hn;
            }
        }

        // wave-local LDS RAW fence (single wave owns this 1 KB tile)
        asm volatile("s_wait_dscnt 0" ::: "memory");

        // -------- rebuild A-layout f16 operand from LDS -------------------
        // A lane l: M = l%16; K = 16hi .. 16hi+15 (contiguous, matches B)
        {
            const _Float16* rowp = myh + n * 32 + 16 * hi;
            v8h lo  = *(const v8h*)rowp;         // K = 16hi .. 16hi+7
            v8h hi8 = *(const v8h*)(rowp + 8);   // K = 16hi+8 .. 16hi+15
            #pragma unroll
            for (int i = 0; i < 8; ++i) { Ah[i] = lo[i]; Ah[8 + i] = hi8[i]; }
        }
    }

    // -------- write h_final: out[b, p*H + j] ------------------------------
    #pragma unroll
    for (int half = 0; half < 2; ++half) {
        #pragma unroll
        for (int r = 0; r < 8; ++r) {
            const int b = b0 + r + 8 * hi;
            const int j = 16 * half + n;
            out[(size_t)b * (P_N * H_N) + p * H_N + j] = hst[half][r];
        }
    }
}

extern "C" void kernel_launch(void* const* d_in, const int* in_sizes, int n_in,
                              void* d_out, int out_size, void* d_ws, size_t ws_size,
                              hipStream_t stream) {
    const float* x    = (const float*)d_in[0];
    const float* W_ih = (const float*)d_in[1];
    const float* W_hh = (const float*)d_in[2];
    const float* b_ih = (const float*)d_in[3];
    const float* b_hh = (const float*)d_in[4];
    float* out = (float*)d_out;

    // 512 batch tiles * 16 processes = 8192 waves; 8 waves per 256-thread block
    dim3 grid((B_TOT / 16) * P_N / 8);   // 1024 blocks
    dim3 block(256);
    gru16_wmma_kernel<<<grid, block, 0, stream>>>(x, W_ih, W_hh, b_ih, b_hh, out);
}
